// CharRNN_6279242186944
// MI455X (gfx1250) — matrix operand order repaired and checked by reading, not compiled, not run on hardware
//
#include <hip/hip_runtime.h>
#include <hip/hip_bf16.h>

// ---------------------------------------------------------------------------
// MI-LSTM CharRNN for MI455X (gfx1250, wave32, WMMA)
//   V=32000, H=256, L=2, B=16, T=256
//   - xh0 = X @ W0 precomputed in a parallel WMMA GEMM (no recurrence in it)
//   - persistent single-workgroup scan does only recurrent GEMMs (U0,W1,U1);
//     weight pointers laundered per-iteration to block LICM -> no spills
//   - logits GEMM: bf16 WMMA, A-tile staged in LDS, HBM-write-bound epilogue
// ---------------------------------------------------------------------------

typedef __bf16        v16bf __attribute__((ext_vector_type(16)));
typedef float         v8f   __attribute__((ext_vector_type(8)));
typedef unsigned int  u32x4 __attribute__((ext_vector_type(4)));

union Frag32B { v16bf v; u32x4 q[2]; };   // 32 bytes: one WMMA A or B fragment

#define VCAB   32000
#define HDIM   256
#define LAY    2
#define BATCH  16
#define TLEN   256
#define G4H    1024   // 4*H

static __device__ __forceinline__ unsigned short f32_to_bf16(float f) {
    union { float f; unsigned int u; } v; v.f = f;
    unsigned int r = v.u + 0x7FFFu + ((v.u >> 16) & 1u);   // round-nearest-even
    return (unsigned short)(r >> 16);
}
static __device__ __forceinline__ float sigm(float x) {
    return 1.0f / (1.0f + __expf(-x));
}
static __device__ __forceinline__ v8f wmma_bf16(v16bf a, v16bf b, v8f c) {
    return __builtin_amdgcn_wmma_f32_16x16x32_bf16(false, a, false, b,
                                                   (short)0, c, false, false);
}
// Defeat LICM: makes the pointer value opaque each time it's called, so loads
// through it cannot be hoisted out of the timestep loop (and then spilled).
static __device__ __forceinline__ const unsigned short* launder_ptr(const unsigned short* p) {
    unsigned long long x = (unsigned long long)p;
    asm volatile("" : "+v"(x));
    return (const unsigned short*)x;
}

// ---------------------------------------------------------------------------
// Kernel 1: W,U [L,H,4H] f32 -> N-major bf16 [L,4H,H]  (so B-fragments are
// two 16B reads per lane matching the CDNA5 B-matrix lane layout)
// ---------------------------------------------------------------------------
__global__ void k_prep_wu(const float* __restrict__ W, const float* __restrict__ U,
                          unsigned short* __restrict__ wt, unsigned short* __restrict__ ut) {
    const int per = LAY * G4H * HDIM;                 // 524288
    int gid = blockIdx.x * blockDim.x + threadIdx.x;
    if (gid >= 2 * per) return;
    const float* src = (gid < per) ? W : U;
    unsigned short* dst = (gid < per) ? wt : ut;
    int i = (gid < per) ? gid : gid - per;
    int l = i / (G4H * HDIM);
    int r = i % (G4H * HDIM);
    int n = r / HDIM;
    int k = r % HDIM;
    dst[i] = f32_to_bf16(src[((size_t)l * HDIM + k) * G4H + n]);
}

// Kernel 2: softmax_w [V,H] f32 -> bf16 (already N-major for the logits GEMM)
__global__ void k_prep_sw(const float* __restrict__ sw, unsigned short* __restrict__ swbf) {
    int gid = blockIdx.x * blockDim.x + threadIdx.x;
    if (gid < VCAB * HDIM) swbf[gid] = f32_to_bf16(sw[gid]);
}

// Kernel 3: embedding gather -> time-major bf16 x [T,B,H]  (row = t*16+b)
__global__ void k_embed(const int* __restrict__ tok, const float* __restrict__ emb,
                        unsigned short* __restrict__ xbf) {
    int gid = blockIdx.x * blockDim.x + threadIdx.x;
    if (gid >= TLEN * BATCH * HDIM) return;
    int t = gid / (BATCH * HDIM);
    int r = gid % (BATCH * HDIM);
    int b = r / HDIM;
    int u = r % HDIM;
    int id = tok[b * TLEN + t];
    xbf[gid] = f32_to_bf16(emb[(size_t)id * HDIM + u]);
}

// ---------------------------------------------------------------------------
// Kernel 4: generic bf16 GEMM  out[M,N] = A[M,256] @ Bt[N,256]^T (+ bias)
// Block = 256 threads = 8 waves; A 16-row tile staged in LDS once per block;
// each wave computes one 16x16 tile; block covers 16(M) x 128(N).
// Used for xh0 (N=1024, bias=null) and logits (N=32000, bias=softmax_b).
// ---------------------------------------------------------------------------
__global__ void __launch_bounds__(256)
k_gemm_bf16(const unsigned short* __restrict__ A,
            const unsigned short* __restrict__ Bt,
            const float* __restrict__ bias,
            float* __restrict__ out, int Ncols)
{
    __shared__ unsigned short Atile[16 * HDIM];   // 8 KB

    const int tid  = threadIdx.x;
    const int lane = tid & 31;
    const int w    = tid >> 5;
    const int lh   = lane & 15;
    const int hi   = lane >> 4;

    const int Mbase = blockIdx.y * 16;
    const int n     = blockIdx.x * 128 + w * 16 + lh;   // output column

    // stage A tile: 16 rows x 256 cols bf16 = 8KB; each thread copies 32B
    {
        int off = tid * 16;                              // element offset
        const unsigned short* src = A + (size_t)Mbase * HDIM + off;
        *(u32x4*)&Atile[off]     = *(const u32x4*)src;
        *(u32x4*)&Atile[off + 8] = *(const u32x4*)(src + 8);
    }
    __syncthreads();

    const unsigned short* brow = Bt + (size_t)n * HDIM;

    v8f acc = (v8f)0.0f;
#pragma unroll
    for (int k0 = 0; k0 < HDIM; k0 += 32) {
        Frag32B a, b;
        const unsigned short* pa = &Atile[lh * HDIM + k0 + hi * 8];
        a.q[0] = *(const u32x4*)pa;
        a.q[1] = *(const u32x4*)(pa + 16);
        const unsigned short* pb = brow + k0 + hi * 8;
        b.q[0] = *(const u32x4*)pb;
        b.q[1] = *(const u32x4*)(pb + 16);
        __builtin_prefetch(pb + 128, 0, 1);   // next K-panel of L2-resident weights
        acc = wmma_bf16(a.v, b.v, acc);
    }

    const float bv = bias ? bias[n] : 0.0f;
#pragma unroll
    for (int jj = 0; jj < 8; ++jj) {
        int row = Mbase + jj + hi * 8;
        out[(size_t)row * Ncols + n] = acc[jj] + bv;
    }
}

// ---------------------------------------------------------------------------
// Kernel 5: persistent MI-LSTM scan. One block, 512 threads = 16 waves.
// Wave w owns hidden units u in [16w,16w+16); it computes its own i/j/f/o
// gate tiles, so activations come straight out of the WMMA accumulators.
//   layer0: xh read from precomputed XH0 (global f32), hh = h0 @ U0 (WMMA)
//   layer1: xh = h0 @ W1 (WMMA),                       hh = h1 @ U1 (WMMA)
// h state: LDS bf16; c state: accumulator-layout registers.
// ---------------------------------------------------------------------------
__global__ void __launch_bounds__(512, 1)
k_scan(const float* __restrict__ XH0,          // [T*B][4H] f32, row = t*16+b
       const unsigned short* __restrict__ wt,  // [L][4H][H] bf16 (only l=1 used)
       const unsigned short* __restrict__ ut,  // [L][4H][H] bf16
       const float* __restrict__ alpha, const float* __restrict__ beta1,
       const float* __restrict__ beta2, const float* __restrict__ bias,
       unsigned short* __restrict__ Xout)      // [B*T][H] bf16 (row = b*T+t)
{
    __shared__ unsigned short hbf[LAY][BATCH * HDIM];   // 16 KB of 320KB WGP LDS

    const int tid  = threadIdx.x;
    const int lane = tid & 31;
    const int w    = tid >> 5;        // wave 0..15 -> hidden-unit block
    const int lh   = lane & 15;
    const int hi   = lane >> 4;       // 0 or 1 (half-wave)

    for (int i = tid; i < LAY * BATCH * HDIM; i += 512) (&hbf[0][0])[i] = 0;
    __syncthreads();

    float creg[LAY][8];
#pragma unroll
    for (int l = 0; l < LAY; ++l)
#pragma unroll
        for (int j = 0; j < 8; ++j) creg[l][j] = 0.0f;

    const int u = w * 16 + lh;        // this lane's hidden-unit column
    float al[LAY][4], bb1[LAY][4], bb2[LAY][4], bbs[LAY][4];
#pragma unroll
    for (int l = 0; l < LAY; ++l)
#pragma unroll
        for (int g = 0; g < 4; ++g) {
            int idx = l * G4H + g * HDIM + u;
            al[l][g]  = alpha[idx];
            bb1[l][g] = beta1[idx];
            bb2[l][g] = beta2[idx];
            bbs[l][g] = bias[idx];
        }

    const int am    = lh;         // A-fragment row for this lane
    const int koff  = hi * 8;     // A-fragment K chunk offset (doc layout)
    const int kbase = hi * 8;     // B-fragment K chunk offset (doc layout)

    for (int t = 0; t < TLEN; ++t) {
#pragma unroll
        for (int l = 0; l < LAY; ++l) {
            // launder per iteration: loads below cannot be hoisted/spilled
            const unsigned short* wl = launder_ptr(wt + (size_t)l * G4H * HDIM);
            const unsigned short* ul = launder_ptr(ut + (size_t)l * G4H * HDIM);

            v8f accX[4], accH[4];
#pragma unroll
            for (int g = 0; g < 4; ++g) { accX[g] = (v8f)0.0f; accH[g] = (v8f)0.0f; }

            if (l == 0) {
                // xh precomputed: read in accumulator layout
                const float* xrow = XH0 + (size_t)t * BATCH * G4H;
#pragma unroll
                for (int g = 0; g < 4; ++g)
#pragma unroll
                    for (int jj = 0; jj < 8; ++jj)
                        accX[g][jj] = xrow[(size_t)(jj + hi * 8) * G4H + g * HDIM + u];
            }

            for (int k0 = 0; k0 < HDIM; k0 += 32) {
                Frag32B aX, aH;
                if (l == 1) {   // layer1 x-input = fresh h0 from LDS
                    const unsigned short* p = &hbf[0][am * HDIM + k0 + koff];
                    aX.q[0] = *(const u32x4*)p;
                    aX.q[1] = *(const u32x4*)(p + 16);
                }
                {
                    const unsigned short* p = &hbf[l][am * HDIM + k0 + koff];
                    aH.q[0] = *(const u32x4*)p;
                    aH.q[1] = *(const u32x4*)(p + 16);
                }
#pragma unroll
                for (int g = 0; g < 4; ++g) {
                    const int n = g * HDIM + u;           // global gate column
                    Frag32B bU;
                    const unsigned short* pu = ul + (size_t)n * HDIM + k0 + kbase;
                    bU.q[0] = *(const u32x4*)pu;
                    bU.q[1] = *(const u32x4*)(pu + 16);
                    accH[g] = wmma_bf16(aH.v, bU.v, accH[g]);
                    if (l == 1) {
                        Frag32B bW;
                        const unsigned short* pw = wl + (size_t)n * HDIM + k0 + kbase;
                        bW.q[0] = *(const u32x4*)pw;
                        bW.q[1] = *(const u32x4*)(pw + 16);
                        accX[g] = wmma_bf16(aX.v, bW.v, accX[g]);
                    }
                }
            }
            __syncthreads();   // all waves done reading hbf[l] (and hbf[0])

            // gate elementwise update, straight from accumulators
#pragma unroll
            for (int jj = 0; jj < 8; ++jj) {
                const int m = jj + hi * 8;   // batch row
                float xi = accX[0][jj], hi_ = accH[0][jj];
                float xj = accX[1][jj], hj_ = accH[1][jj];
                float xf = accX[2][jj], hf_ = accH[2][jj];
                float xo = accX[3][jj], ho_ = accH[3][jj];
                float gi = al[l][0]*xi*hi_ + bb1[l][0]*xi + bb2[l][0]*hi_ + bbs[l][0];
                float gj = al[l][1]*xj*hj_ + bb1[l][1]*xj + bb2[l][1]*hj_ + bbs[l][1];
                float gf = al[l][2]*xf*hf_ + bb1[l][2]*xf + bb2[l][2]*hf_ + bbs[l][2];
                float go = al[l][3]*xo*ho_ + bb1[l][3]*xo + bb2[l][3]*ho_ + bbs[l][3];
                float c = creg[l][jj];
                c = c * sigm(gf + 1.0f) + sigm(gi) * tanhf(gj);   // FORGET_BIAS=1
                float h = tanhf(c) * sigm(go);
                creg[l][jj] = c;
                unsigned short hb = f32_to_bf16(h);
                hbf[l][m * HDIM + u] = hb;
                if (l == 1)      // scan output row index = b*T + t
                    Xout[((size_t)m * TLEN + t) * HDIM + u] = hb;
            }
            __syncthreads();   // new h visible before anyone reads it
        }
    }
}

// ---------------------------------------------------------------------------
extern "C" void kernel_launch(void* const* d_in, const int* in_sizes, int n_in,
                              void* d_out, int out_size, void* d_ws, size_t ws_size,
                              hipStream_t stream) {
    const int*   tokens = (const int*)  d_in[0];   // [B,T]
    const float* emb    = (const float*)d_in[1];   // [V,H]
    const float* W      = (const float*)d_in[2];   // [L,H,4H]
    const float* U      = (const float*)d_in[3];   // [L,H,4H]
    const float* alpha  = (const float*)d_in[4];   // [L,4H]
    const float* beta1  = (const float*)d_in[5];
    const float* beta2  = (const float*)d_in[6];
    const float* bias   = (const float*)d_in[7];
    const float* sw     = (const float*)d_in[8];   // [V,H]
    const float* sb     = (const float*)d_in[9];   // [V]
    float* out = (float*)d_out;                    // [B*T, V]

    // workspace carve-up: bf16 buffers then f32 XH0 (~39 MB total)
    unsigned short* wt   = (unsigned short*)d_ws;              // 524288 elems
    unsigned short* ut   = wt   + (size_t)LAY * G4H * HDIM;    // 524288
    unsigned short* swbf = ut   + (size_t)LAY * G4H * HDIM;    // 8192000
    unsigned short* xbf  = swbf + (size_t)VCAB * HDIM;         // 1048576
    unsigned short* Xout = xbf  + (size_t)TLEN * BATCH * HDIM; // 1048576
    float* XH0 = (float*)(Xout + (size_t)BATCH * TLEN * HDIM); // 4096*1024 f32

    {   // weight transpose+convert
        int n = 2 * LAY * G4H * HDIM;
        k_prep_wu<<<(n + 255) / 256, 256, 0, stream>>>(W, U, wt, ut);
    }
    {   // softmax weights to bf16
        int n = VCAB * HDIM;
        k_prep_sw<<<(n + 255) / 256, 256, 0, stream>>>(sw, swbf);
    }
    {   // embedding gather
        int n = TLEN * BATCH * HDIM;
        k_embed<<<(n + 255) / 256, 256, 0, stream>>>(tokens, emb, xbf);
    }
    {   // xh0 = X @ W0 : parallel WMMA GEMM [4096,256]x[256,1024]
        dim3 grid(G4H / 128, (TLEN * BATCH) / 16);
        k_gemm_bf16<<<grid, 256, 0, stream>>>(xbf, wt, nullptr, XH0, G4H);
    }
    // persistent recurrence: one workgroup, 16 waves
    k_scan<<<1, 512, 0, stream>>>(XH0, wt, ut, alpha, beta1, beta2, bias, Xout);
    {   // logits GEMM [4096,256]x[256,32000] + bias
        dim3 grid(VCAB / 128, (TLEN * BATCH) / 16);
        k_gemm_bf16<<<grid, 256, 0, stream>>>(Xout, swbf, sb, out, VCAB);
    }
}
